// Decoder_14465449853643
// MI455X (gfx1250) — matrix operand order repaired and checked by reading, not compiled, not run on hardware
//
#include <hip/hip_runtime.h>
#include <math.h>

// ---------------------------------------------------------------------------
// LSTM caption decoder for MI455X (gfx1250), wave32 + WMMA bf16 + LDS staging.
//   E = H = 512, V = 32000, B = 64, T = 20 (T+1 = 21 LSTM steps)
// ---------------------------------------------------------------------------

#define E_DIM 512
#define H_DIM 512
#define V_DIM 32000
#define B_DIM 64
#define T_DIM 20
#define S_DIM (T_DIM + 1)        // 21 LSTM steps
#define G_DIM (4 * H_DIM)        // 2048 gate width
#define HSLOT (S_DIM + 1)        // 22 hidden slots per batch (slot0 = h_{-1} = 0)

typedef __bf16 bf16_t;
typedef __attribute__((ext_vector_type(16))) __bf16       v16bf;
typedef __attribute__((ext_vector_type(8)))  float        v8f;
typedef __attribute__((ext_vector_type(8)))  unsigned int v8u;

union ABFrag { v8u u; v16bf b; };

__device__ __forceinline__ unsigned short f32_to_bf16(float f) {
    unsigned int u = __float_as_uint(f);
    u += 0x7FFFu + ((u >> 16) & 1u);          // round-to-nearest-even
    return (unsigned short)(u >> 16);
}

// A fragment from row-major bf16: per-lane K striping {hi*8 + (0..7 | 16..23)}
__device__ __forceinline__ void load_a_frag(ABFrag& a, const unsigned short* Arow,
                                            int kt, int hi) {
    int kbase = (kt << 5) + (hi << 3);
#pragma unroll
    for (int v = 0; v < 8; ++v) {
        int k0 = kbase + ((v < 4) ? (v << 1) : (16 + ((v - 4) << 1)));
        a.u[v] = *(const unsigned int*)(Arow + k0);
    }
}

// ---------------------------------------------------------------------------
// Pack fp32 weight W[N][K] (row-major) into bf16 B-tiles matching the
// V_WMMA_*_16x16x32 B-matrix VGPR striping (32x16 tile, wave32):
//   VGPR v (0..7), lane l: n = nt*16 + (l&15); k0 = kt*32 + (l>=16?8:0) + (v<4 ? 2v : 16+2(v-4))
// Tile order: (nt * Ktiles + kt), within tile lane-major, 8 dwords per lane.
// ---------------------------------------------------------------------------
__global__ void pack_b_bf16(const float* __restrict__ W, unsigned int* __restrict__ out,
                            int Ktiles, int K, int total) {
    int idx = blockIdx.x * blockDim.x + threadIdx.x;
    if (idx >= total) return;
    int v    = idx & 7;
    int lane = (idx >> 3) & 31;
    int tile = idx >> 8;
    int kt = tile % Ktiles;
    int nt = tile / Ktiles;
    int n  = (nt << 4) + (lane & 15);
    int k0 = (kt << 5) + ((lane >> 4) << 3) + ((v < 4) ? (v << 1) : (16 + ((v - 4) << 1)));
    float w0 = W[(size_t)n * K + k0];
    float w1 = W[(size_t)n * K + k0 + 1];
    out[idx] = (unsigned int)f32_to_bf16(w0) | ((unsigned int)f32_to_bf16(w1) << 16);
}

// combined LSTM bias: b_ih + b_hh
__global__ void bias_combine(const float* __restrict__ a, const float* __restrict__ b,
                             float* __restrict__ o) {
    int i = blockIdx.x * blockDim.x + threadIdx.x;
    if (i < G_DIM) o[i] = a[i] + b[i];
}

// xs[b][t][e] (bf16): t==0 -> features, else embedding gather of captions[b][t-1]
__global__ void build_xs(const float* __restrict__ features, const int* __restrict__ captions,
                         const float* __restrict__ emb, unsigned short* __restrict__ xs,
                         int total) {
    int idx = blockIdx.x * blockDim.x + threadIdx.x;
    if (idx >= total) return;
    int e = idx & (E_DIM - 1);
    int r = idx >> 9;
    int b = r / S_DIM, t = r % S_DIM;
    float v = (t == 0) ? features[(size_t)b * E_DIM + e]
                       : emb[(size_t)captions[b * T_DIM + (t - 1)] * E_DIM + e];
    xs[idx] = f32_to_bf16(v);
}

// zero c[64][512] and hidden slot 0 (h_{-1}) for every batch
__global__ void zero_init(float* __restrict__ c, unsigned int* __restrict__ hs_dw) {
    int idx = blockIdx.x * blockDim.x + threadIdx.x;
    if (idx < B_DIM * H_DIM) c[idx] = 0.f;
    if (idx < B_DIM * (H_DIM / 2)) {
        int b = idx >> 8, w = idx & 255;
        hs_dw[(size_t)b * HSLOT * (H_DIM / 2) + w] = 0u;
    }
}

// ---------------------------------------------------------------------------
// bf16 WMMA GEMM with cooperative LDS B-panel staging, software pipelined.
//   Block = 4 waves; wave w owns M-tile (mb*4+w). Panel = NT N-tiles (NT=4|8)
//   shared through LDS (double-buffered, 1 barrier/k-tile). Source order:
//     compute(kt) -> ds_store B(kt+1) -> issue global B(kt+2), A(kt+1) -> barrier
//   so the WMMAs only wait on this k-tile's ds_loads + an A fragment that was
//   issued a full iteration earlier (partial loadcnt wait, no full drain).
//   A: row-major bf16, row index = (m/mdiv)*mmul + (m%mdiv) + moff, stride K.
//   Grids are exact multiples; no bounds checks. blockDim.x = 128.
// ---------------------------------------------------------------------------
template <int NT>
__global__ void gemm_wmma(const unsigned short* __restrict__ A,
                          const unsigned int*   __restrict__ Bp,
                          const float* __restrict__ bias,
                          const float* __restrict__ Cin, int ldcin,
                          float* __restrict__ Cout, int ldcout,
                          int K, int Ktiles, int Ntiles,
                          int mdiv, int mmul, int moff) {
    constexpr int NSTG = NT / 4;                       // staging slots per thread
    __shared__ __align__(32) unsigned int ldsB[2][NT * 256];

    int wid  = threadIdx.x >> 5;
    int lane = threadIdx.x & 31;
    int nbg  = Ntiles / NT;
    int mb   = blockIdx.x / nbg;
    int nt0  = (blockIdx.x % nbg) * NT;
    int mt   = (mb << 2) + wid;

    int lo = lane & 15;
    int hi = lane >> 4;
    int mloc = (mt << 4) + lo;
    int arow = (mloc / mdiv) * mmul + (mloc % mdiv) + moff;
    const unsigned short* Arow = A + (size_t)arow * K;

    // this thread stages tiles (wid + s*4), s = 0..NSTG-1
    const unsigned int* Bst[NSTG];
#pragma unroll
    for (int s = 0; s < NSTG; ++s)
        Bst[s] = Bp + (size_t)(nt0 + wid + s * 4) * Ktiles * 256 + lane * 8;

    v8f acc[NT];
    if (Cin) {
#pragma unroll
        for (int j = 0; j < NT; ++j) {
            int n = ((nt0 + j) << 4) + lo;
#pragma unroll
            for (int r = 0; r < 8; ++r) {
                int m = (mt << 4) + r + (hi << 3);
                acc[j][r] = Cin[(size_t)m * ldcin + n];
            }
        }
    } else {
#pragma unroll
        for (int j = 0; j < NT; ++j) acc[j] = (v8f)0.f;
    }

    // ---- prologue: B(0) -> lds[0]; B(1) -> breg; A(0) -> a -----------------
    v8u breg[NSTG];
#pragma unroll
    for (int s = 0; s < NSTG; ++s) breg[s] = *(const v8u*)Bst[s];
#pragma unroll
    for (int s = 0; s < NSTG; ++s)
        *(v8u*)&ldsB[0][(wid + s * 4) * 256 + lane * 8] = breg[s];
    if (Ktiles > 1) {
#pragma unroll
        for (int s = 0; s < NSTG; ++s) breg[s] = *(const v8u*)(Bst[s] + 256);
    }
    ABFrag a;
    load_a_frag(a, Arow, 0, hi);
    __syncthreads();

    int buf = 0;
    for (int kt = 0; kt < Ktiles; ++kt) {
        // compute: lds[buf] holds B(kt); 'a' holds A(kt) (issued last iter)
#pragma unroll
        for (int j = 0; j < NT; ++j) {
            ABFrag b;
            b.u = *(const v8u*)&ldsB[buf][j * 256 + lane * 8];
            acc[j] = __builtin_amdgcn_wmma_f32_16x16x32_bf16(
                false, a.b, false, b.b, (short)0, acc[j], false, false);
        }

        if (kt + 1 < Ktiles) {
            // stage B(kt+1) regs -> lds[buf^1]
#pragma unroll
            for (int s = 0; s < NSTG; ++s)
                *(v8u*)&ldsB[buf ^ 1][(wid + s * 4) * 256 + lane * 8] = breg[s];
            // issue B(kt+2) and A(kt+1); consumed only after the next barrier
            if (kt + 2 < Ktiles) {
#pragma unroll
                for (int s = 0; s < NSTG; ++s) {
                    const unsigned int* nb = Bst[s] + (size_t)(kt + 2) * 256;
                    breg[s] = *(const v8u*)nb;
                    if (s == 0 && kt + 3 < Ktiles) __builtin_prefetch(nb + 256, 0, 1);
                }
            }
            load_a_frag(a, Arow, kt + 1, hi);
        }
        __syncthreads();
        buf ^= 1;
    }

#pragma unroll
    for (int j = 0; j < NT; ++j) {
        int n = ((nt0 + j) << 4) + lo;
        float bv = bias ? bias[n] : 0.f;
#pragma unroll
        for (int r = 0; r < 8; ++r) {
            int m = (mt << 4) + r + (hi << 3);
            Cout[(size_t)m * ldcout + n] = acc[j][r] + bv;
        }
    }
}

// LSTM pointwise: gates[64][2048] (i,f,g,o) -> c update, h_s -> hs slot s+1 (bf16)
__global__ void lstm_pointwise(const float* __restrict__ gates, float* __restrict__ c,
                               unsigned short* __restrict__ hs, int s) {
    int idx = blockIdx.x * blockDim.x + threadIdx.x;   // B*H = 32768 threads
    int b = idx >> 9, j = idx & (H_DIM - 1);
    const float* g = gates + (size_t)b * G_DIM;
    float i = 1.f / (1.f + __expf(-g[j]));
    float f = 1.f / (1.f + __expf(-g[H_DIM + j]));
    float t = tanhf(g[2 * H_DIM + j]);
    float o = 1.f / (1.f + __expf(-g[3 * H_DIM + j]));
    float cn = f * c[idx] + i * t;
    c[idx] = cn;
    float h = o * tanhf(cn);
    hs[(size_t)(b * HSLOT + s + 1) * H_DIM + j] = f32_to_bf16(h);
}

// per-row max + 1/sum(exp) over V=32000
__global__ void row_softmax_stats(const float* __restrict__ logits, float2* __restrict__ st) {
    __shared__ float red[256];
    int row = blockIdx.x;
    const float* x = logits + (size_t)row * V_DIM;
    float m = -1e30f;
    for (int i = threadIdx.x; i < V_DIM; i += 256) m = fmaxf(m, x[i]);
    red[threadIdx.x] = m; __syncthreads();
    for (int s = 128; s > 0; s >>= 1) {
        if (threadIdx.x < s) red[threadIdx.x] = fmaxf(red[threadIdx.x], red[threadIdx.x + s]);
        __syncthreads();
    }
    m = red[0]; __syncthreads();
    float sum = 0.f;
    for (int i = threadIdx.x; i < V_DIM; i += 256) sum += __expf(x[i] - m);
    red[threadIdx.x] = sum; __syncthreads();
    for (int s = 128; s > 0; s >>= 1) {
        if (threadIdx.x < s) red[threadIdx.x] += red[threadIdx.x + s];
        __syncthreads();
    }
    if (threadIdx.x == 0) st[row] = make_float2(m, 1.f / red[0]);
}

// normalize in place, float4-vectorized (V % 4 == 0)
__global__ void softmax_norm(float* __restrict__ logits, const float2* __restrict__ st) {
    int idx = blockIdx.x * blockDim.x + threadIdx.x;   // over float4 elements
    int row = idx / (V_DIM / 4);
    float2 s = st[row];
    float4* p = (float4*)logits + idx;
    float4 v = *p;
    v.x = __expf(v.x - s.x) * s.y;
    v.y = __expf(v.y - s.x) * s.y;
    v.z = __expf(v.z - s.x) * s.y;
    v.w = __expf(v.w - s.x) * s.y;
    *p = v;
}

// ---------------------------------------------------------------------------
extern "C" void kernel_launch(void* const* d_in, const int* in_sizes, int n_in,
                              void* d_out, int out_size, void* d_ws, size_t ws_size,
                              hipStream_t stream) {
    const float* features = (const float*)d_in[0];
    const int*   captions = (const int*)d_in[1];
    /* d_in[2] = lenghts: unused by the reference */
    const float* emb   = (const float*)d_in[3];
    const float* w_ih  = (const float*)d_in[4];
    const float* w_hh  = (const float*)d_in[5];
    const float* b_ih  = (const float*)d_in[6];
    const float* b_hh  = (const float*)d_in[7];
    const float* fc_w  = (const float*)d_in[8];
    const float* fc_b  = (const float*)d_in[9];
    float* out = (float*)d_out;

    // workspace carve-out (~51.5 MB), 256B aligned
    char* p = (char*)d_ws;
    auto carve = [&](size_t bytes) { char* r = p; p += (bytes + 255) & ~(size_t)255; return r; };
    unsigned int*   WihP    = (unsigned int*)  carve((size_t)E_DIM * G_DIM * 2);   // 2 MB
    unsigned int*   WhhP    = (unsigned int*)  carve((size_t)H_DIM * G_DIM * 2);   // 2 MB
    unsigned int*   FcwP    = (unsigned int*)  carve((size_t)H_DIM * V_DIM * 2);   // 32 MB
    unsigned short* xs      = (unsigned short*)carve((size_t)B_DIM * S_DIM * E_DIM * 2);
    float*          cbias   = (float*)         carve((size_t)G_DIM * 4);
    float*          gates_x = (float*)         carve((size_t)B_DIM * S_DIM * G_DIM * 4); // 11 MB
    float*          gates_t = (float*)         carve((size_t)B_DIM * G_DIM * 4);
    float*          cbuf    = (float*)         carve((size_t)B_DIM * H_DIM * 4);
    unsigned short* hs      = (unsigned short*)carve((size_t)B_DIM * HSLOT * H_DIM * 2);
    float2*         stats   = (float2*)        carve((size_t)B_DIM * T_DIM * 8);

    const int KT = E_DIM / 32;               // 16 k-tiles (E == H == 512)

    // 1) pack weights into WMMA B-tile layout (bf16)
    {
        int tot = KT * (G_DIM / 16) * 256;   // 524288
        pack_b_bf16<<<(tot + 255) / 256, 256, 0, stream>>>(w_ih, WihP, KT, E_DIM, tot);
        pack_b_bf16<<<(tot + 255) / 256, 256, 0, stream>>>(w_hh, WhhP, KT, H_DIM, tot);
        int totf = KT * (V_DIM / 16) * 256;  // 8192000
        pack_b_bf16<<<(totf + 255) / 256, 256, 0, stream>>>(fc_w, FcwP, KT, H_DIM, totf);
    }
    // 2) combined bias, xs assembly, state zeroing
    bias_combine<<<G_DIM / 256, 256, 0, stream>>>(b_ih, b_hh, cbias);
    {
        int tot = B_DIM * S_DIM * E_DIM;     // 688128
        build_xs<<<(tot + 255) / 256, 256, 0, stream>>>(features, captions, emb, xs, tot);
    }
    zero_init<<<(B_DIM * H_DIM) / 256, 256, 0, stream>>>(cbuf, (unsigned int*)hs);

    // 3) hoisted input GEMM: gates_x[1344][2048] = xs @ W_ih^T + (b_ih + b_hh)
    {
        int blocks = (B_DIM * S_DIM / 64) * ((G_DIM / 16) / 8);   // 21 * 16 = 336
        gemm_wmma<8><<<blocks, 128, 0, stream>>>(
            xs, WihP, cbias, nullptr, 0, gates_x, G_DIM,
            E_DIM, KT, G_DIM / 16, 1 << 30, 0, 0);
    }

    // 4) recurrence: 21 sequential steps of [64x512] @ [512x2048] + pointwise
    //    NT=4 keeps 32 blocks/step for latency-bound parallelism.
    for (int s = 0; s < S_DIM; ++s) {
        // gates_t = gates_x[:, s, :] + h_{s-1} @ W_hh^T   (h_{s-1} = hs slot s)
        gemm_wmma<4><<<32, 128, 0, stream>>>(
            hs, WhhP, nullptr, gates_x + (size_t)s * G_DIM, S_DIM * G_DIM,
            gates_t, G_DIM, H_DIM, KT, G_DIM / 16, 1, HSLOT, s);
        lstm_pointwise<<<(B_DIM * H_DIM) / 256, 256, 0, stream>>>(gates_t, cbuf, hs, s);
    }

    // 5) FC: logits[1280][32000] = h_{1..20} @ fc_w^T + fc_b  (row (b,t) -> slot t+2)
    {
        int blocks = (B_DIM * T_DIM / 64) * ((V_DIM / 16) / 8);   // 20 * 250 = 5000
        gemm_wmma<8><<<blocks, 128, 0, stream>>>(
            hs, FcwP, fc_b, nullptr, 0, out, V_DIM,
            H_DIM, KT, V_DIM / 16, T_DIM, HSLOT, 2);
    }

    // 6) softmax over V per row
    row_softmax_stats<<<B_DIM * T_DIM, 256, 0, stream>>>(out, stats);
    {
        int tot4 = B_DIM * T_DIM * (V_DIM / 4);                  // 10,240,000
        softmax_norm<<<tot4 / 256, 256, 0, stream>>>(out, stats);
    }
}
